// SimpleCrossMultiHeadAttention_7816840478958
// MI455X (gfx1250) — compile-verified, matrix operand order checked by reading, not run on hardware
//
#include <hip/hip_runtime.h>
#include <math.h>

typedef __attribute__((ext_vector_type(16))) _Float16 v16h;
typedef __attribute__((ext_vector_type(8)))  float    v8f;

#define NHEAD 8
#define DMODEL 256
#define DHEAD 32
#define SHID 64
#define NB 4
#define NEGV (-1.0e6f)
#define KPAD 68   // SHID + 4 float pad: keeps 16B alignment for B128, spreads banks

// K-index a lane's i-th f16 element maps to in the 16x32 A / 32x16 B WMMA tile.
__device__ __forceinline__ int kmap16(int i, int lane) {
  return ((i < 8) ? i : i + 8) + ((lane & 16) ? 8 : 0);
}

// gfx1250 async global->LDS copy (ASYNCcnt-tracked), 16 bytes per lane.
// LDS byte address = low 32 bits of the generic pointer.
__device__ __forceinline__ void async_copy_b128(void* lds_dst, const void* gsrc) {
  unsigned l = (unsigned)(size_t)lds_dst;
  asm volatile("global_load_async_to_lds_b128 %0, %1, off"
               :: "v"(l), "v"(gsrc) : "memory");
}
__device__ __forceinline__ void async_wait0() {
  asm volatile("s_wait_asynccnt 0x0" ::: "memory");
}

// F[bh, l, 0:SHID] = X[b, l, head*DHEAD : +DHEAD] @ W[DHEAD, SHID]
// One wave = one 16x16 output tile; K=32 covered by a single WMMA.
__global__ void proj_kernel(const float* __restrict__ X,
                            const float* __restrict__ W,
                            float* __restrict__ F, int L) {
  const int lane = threadIdx.x & 31;
  const int wave = threadIdx.x >> 5;          // 0..3 -> N tile of SHID=64
  const int tile = blockIdx.x;                // 16-row tile over BH*L rows
  const int n    = lane & 15;
  const int r    = tile * 16 + n;
  const int bh   = r / L;
  const int l    = r % L;
  const int b    = bh / NHEAD;
  const int head = bh % NHEAD;
  const float* arow = X + ((size_t)b * L + l) * DMODEL + head * DHEAD;
  const int col  = wave * 16 + n;

  v16h a, w;
#pragma unroll
  for (int i = 0; i < 16; ++i) {
    int k = kmap16(i, lane);
    a[i] = (_Float16)arow[k];
    w[i] = (_Float16)W[k * SHID + col];
  }
  v8f c = {};
  c = __builtin_amdgcn_wmma_f32_16x16x32_f16(false, a, false, w, (short)0, c,
                                             false, false);
  const int moff = (lane & 16) ? 8 : 0;
#pragma unroll
  for (int v = 0; v < 8; ++v) {
    int r2 = tile * 16 + v + moff;
    F[((size_t)(r2 / L) * L + (r2 % L)) * SHID + col] = c[v];
  }
}

// Fused additive-attention core for one (bh, 16-query tile), 4 waves/block:
// async-staged tiles -> tanh scores (VALU, waves split Lk) -> mask ->
// shuffle-tree softmax -> P@V via f16 WMMA (waves 0/1 own the two N tiles).
__global__ void attn_kernel(const float* __restrict__ QF,    // [BH, Lq, SHID]
                            const float* __restrict__ KF,    // [BH, Lk, SHID]
                            const float* __restrict__ V,     // [B, Lk, DMODEL]
                            const float* __restrict__ wv,    // [SHID]
                            const int*   __restrict__ valid, // [B]
                            float* __restrict__ AO,          // [B, Lq, DMODEL]
                            int Lq, int Lk) {
  __shared__ __align__(16) float sQF[16 * SHID];
  __shared__ __align__(16) float sKF[4][16 * KPAD];  // per-wave kf chunk
  __shared__ __align__(16) float sS[16 * 256];       // max Lk
  __shared__ float sWv[SHID];

  const int tid  = threadIdx.x;            // 0..127
  const int lane = tid & 31;
  const int wave = tid >> 5;               // 0..3
  const int qt   = blockIdx.x;
  const int bh   = blockIdx.y;
  const int b    = bh / NHEAD;
  const int head = bh % NHEAD;
  const int vl   = valid[b];
  const int n    = lane & 15;
  const int moff = (lane & 16) ? 8 : 0;

  // ---- async-stage qf tile (16 x SHID) + wv ----
  {
    const float* gq = QF + ((size_t)bh * Lq + qt * 16) * SHID;
    int e = tid * 8;
    async_copy_b128(&sQF[e], gq + e);
    async_copy_b128(&sQF[e + 4], gq + e + 4);
    for (int i = tid; i < SHID; i += 128) sWv[i] = wv[i];
    async_wait0();
  }
  __syncthreads();

  // ---- scores: s[m,k] = sum_h wv[h] * tanh(qf[m,h] + kf[k,h]) ----
  float* skf = &sKF[wave][0];
  for (int kc = wave; kc < Lk / 16; kc += 4) {
    // async-stage this wave's kf chunk [16 x SHID] into padded LDS
#pragma unroll
    for (int it = 0; it < 8; ++it) {
      int u = it * 32 + lane;
      int row = u >> 4, c4 = (u & 15) * 4;
      async_copy_b128(&skf[row * KPAD + c4],
                      KF + ((size_t)bh * Lk + kc * 16 + row) * SHID + c4);
    }
    async_wait0();   // producer == consumer wave: no block barrier needed

    const int k = kc * 16 + n;
    const float* kfr = &skf[n * KPAD];
#pragma unroll
    for (int v = 0; v < 8; ++v) {
      const int m = v + moff;
      float s = 0.f;
#pragma unroll 8
      for (int h = 0; h < SHID; ++h)
        s += sWv[h] * tanhf(sQF[m * SHID + h] + kfr[h]);
      sS[m * Lk + k] = (k < vl) ? s : NEGV;
    }
  }
  __syncthreads();

  // ---- softmax: 16 rows x 8 lanes, shfl_xor tree reductions ----
  {
    const int row = tid >> 3, sub = tid & 7;  // 8-lane groups stay in a wave
    float* r = &sS[row * Lk];
    float mx = -3.0e38f;
    for (int k = sub; k < Lk; k += 8) mx = fmaxf(mx, r[k]);
    mx = fmaxf(mx, __shfl_xor(mx, 1));
    mx = fmaxf(mx, __shfl_xor(mx, 2));
    mx = fmaxf(mx, __shfl_xor(mx, 4));
    float sum = 0.f;
    for (int k = sub; k < Lk; k += 8) {
      float e = __expf(r[k] - mx);
      r[k] = e;
      sum += e;
    }
    sum += __shfl_xor(sum, 1);
    sum += __shfl_xor(sum, 2);
    sum += __shfl_xor(sum, 4);
    float inv = 1.0f / sum;
    for (int k = sub; k < Lk; k += 8) r[k] *= inv;
  }
  __syncthreads();

  // ---- out[16 x 32] = P[16 x Lk] @ V_head[Lk x 32]; wave w<2 owns N tile w --
  if (wave < 2) {
    const int nt = wave;
    v8f acc = {};
    for (int kc = 0; kc < Lk / 32; ++kc) {
      v16h a, bt;
#pragma unroll
      for (int i = 0; i < 16; ++i) {
        int kk = kc * 32 + kmap16(i, lane);
        a[i]  = (_Float16)sS[n * Lk + kk];
        bt[i] = (_Float16)V[((size_t)b * Lk + kk) * DMODEL + head * DHEAD +
                            nt * 16 + n];
      }
      acc = __builtin_amdgcn_wmma_f32_16x16x32_f16(false, a, false, bt,
                                                   (short)0, acc, false, false);
    }
#pragma unroll
    for (int v = 0; v < 8; ++v) {
      int q = qt * 16 + v + moff;
      AO[((size_t)b * Lq + q) * DMODEL + head * DHEAD + nt * 16 + n] = acc[v];
    }
  }
}

// Out[Rows, 256] = A[Rows, 256] @ Wo[256, 256]
__global__ void outproj_kernel(const float* __restrict__ A,
                               const float* __restrict__ Wo,
                               float* __restrict__ Out, int Rows) {
  const int lane = threadIdx.x & 31;
  const int wave = threadIdx.x >> 5;
  const int nt   = blockIdx.y * 4 + wave;   // 0..15
  const int rt   = blockIdx.x;
  const int n    = lane & 15;
  const int col  = nt * 16 + n;
  const int moff = (lane & 16) ? 8 : 0;
  const float* arow = A + (size_t)(rt * 16 + n) * DMODEL;

  v8f acc = {};
  for (int kc = 0; kc < DMODEL / 32; ++kc) {
    __builtin_prefetch(&Wo[(size_t)((kc + 1) * 32 % DMODEL) * DMODEL + col], 0, 1);
    v16h a, w;
#pragma unroll
    for (int i = 0; i < 16; ++i) {
      int k = kc * 32 + kmap16(i, lane);
      a[i] = (_Float16)arow[k];
      w[i] = (_Float16)Wo[(size_t)k * DMODEL + col];
    }
    acc = __builtin_amdgcn_wmma_f32_16x16x32_f16(false, a, false, w, (short)0,
                                                 acc, false, false);
  }
#pragma unroll
  for (int v = 0; v < 8; ++v)
    Out[(size_t)(rt * 16 + v + moff) * DMODEL + col] = acc[v];
}

extern "C" void kernel_launch(void* const* d_in, const int* in_sizes, int n_in,
                              void* d_out, int out_size, void* d_ws,
                              size_t ws_size, hipStream_t stream) {
  const float* q_l  = (const float*)d_in[0];
  const float* k_r  = (const float*)d_in[1];
  const float* v_r  = (const float*)d_in[2];
  const float* q_r  = (const float*)d_in[3];
  const float* k_l  = (const float*)d_in[4];
  const float* v_l  = (const float*)d_in[5];
  const float* Wq_l = (const float*)d_in[6];
  const float* Wk_l = (const float*)d_in[7];
  const float* wv_l = (const float*)d_in[8];
  const float* Wo_l = (const float*)d_in[9];
  const float* Wq_r = (const float*)d_in[10];
  const float* Wk_r = (const float*)d_in[11];
  const float* wv_r = (const float*)d_in[12];
  const float* Wo_r = (const float*)d_in[13];
  const int*   vlr  = (const int*)d_in[14];
  const int*   vll  = (const int*)d_in[15];
  float* out = (float*)d_out;

  float* ws   = (float*)d_ws;
  float* qf_l = ws;                         // 32*128*64
  float* kf_l = qf_l + 32 * 128 * 64;       // 32*256*64
  float* qf_r = kf_l + 32 * 256 * 64;       // 32*256*64
  float* kf_r = qf_r + 32 * 256 * 64;       // 32*128*64
  float* ao_l = kf_r + 32 * 128 * 64;       // 4*128*256
  float* ao_r = ao_l + 4 * 128 * 256;       // 4*256*256

  // projections (qf/kf) — one WMMA per 16x16 tile, K=32 in one instruction
  proj_kernel<<<dim3(32 * 128 / 16), 128, 0, stream>>>(q_l, Wq_l, qf_l, 128);
  proj_kernel<<<dim3(32 * 256 / 16), 128, 0, stream>>>(k_r, Wk_l, kf_l, 256);
  proj_kernel<<<dim3(32 * 256 / 16), 128, 0, stream>>>(q_r, Wq_r, qf_r, 256);
  proj_kernel<<<dim3(32 * 128 / 16), 128, 0, stream>>>(k_l, Wk_r, kf_r, 128);

  // fused tanh-score + mask + softmax + P@V  (4 waves per 16-query tile)
  attn_kernel<<<dim3(128 / 16, 32), 128, 0, stream>>>(qf_l, kf_l, v_r, wv_l,
                                                      vlr, ao_l, 128, 256);
  attn_kernel<<<dim3(256 / 16, 32), 128, 0, stream>>>(qf_r, kf_r, v_l, wv_r,
                                                      vll, ao_r, 256, 128);

  // merged-heads @ Wo -> d_out (left then right, flat concat)
  outproj_kernel<<<dim3(4 * 128 / 16, 4), 128, 0, stream>>>(ao_l, Wo_l, out,
                                                            4 * 128);
  outproj_kernel<<<dim3(4 * 256 / 16, 4), 128, 0, stream>>>(
      ao_r, Wo_r, out + 4 * 128 * 256, 4 * 256);
}
